// PEMultiHeadAttention_81578608820251
// MI455X (gfx1250) — compile-verified, matrix operand order checked by reading, not run on hardware
//
#include <hip/hip_runtime.h>

// ---------------------------------------------------------------------------
// PE multi-head attention for MI455X (gfx1250, wave32, WMMA).
// B=4, N=M=2048, D=512, H=8, DH=64.
//   proj_kernel<DUAL,TR> x3 : Q,K = X@W + E@Wp + b (bf16, head-split),
//                             V^T = X@Wv + b (bf16, (b,h,dh,m))
//   attn_kernel             : per (b,h,16 q-rows): QK^T via v_wmma bf16,
//                             full-row fp32 softmax in 132KB dynamic LDS,
//                             probs re-packed bf16 in-place so the PV loop is
//                             pure ds/global b128 loads + WMMA (8-wave split-K
//                             with LDS partial reduction).
// ---------------------------------------------------------------------------

typedef __attribute__((ext_vector_type(16))) __bf16        v16bf;
typedef __attribute__((ext_vector_type(8)))  __bf16        v8bf;
typedef __attribute__((ext_vector_type(8)))  float         v8f;
typedef __attribute__((ext_vector_type(4)))  float         fvec4;
typedef __attribute__((ext_vector_type(2)))  float         fvec2;
typedef __attribute__((ext_vector_type(4)))  unsigned int  uvec4;

constexpr int cB = 4, cN = 2048, cM = 2048, cD = 512, cH = 8, cDH = 64;

__device__ __forceinline__ unsigned short f2bf_u(float f) {
  unsigned u = __builtin_bit_cast(unsigned, f);
  unsigned r = u + 0x7FFFu + ((u >> 16) & 1u);   // round-to-nearest-even
  return (unsigned short)(r >> 16);
}
__device__ __forceinline__ v8f vzero8() {
  v8f z;
#pragma unroll
  for (int i = 0; i < 8; ++i) z[i] = 0.0f;
  return z;
}
// Assemble one 16-bit 16x32 A/B fragment (per ISA §7.12.2 layout): lane l holds
// row/col l&15, K-groups {kg*8..+7, 16+kg*8..+7}, kg = l>>4. Two 16B loads.
__device__ __forceinline__ v16bf load_bf16x16(const unsigned short* p0,
                                              const unsigned short* p1) {
  v8bf lo = __builtin_bit_cast(v8bf, *(const uvec4*)p0);
  v8bf hi = __builtin_bit_cast(v8bf, *(const uvec4*)p1);
  v16bf r;
#pragma unroll
  for (int i = 0; i < 8; ++i) { r[i] = lo[i]; r[8 + i] = hi[i]; }
  return r;
}
__device__ __forceinline__ v8f wmma_bf(v16bf a, v16bf b, v8f c) {
  return __builtin_amdgcn_wmma_f32_16x16x32_bf16(false, a, false, b,
                                                 (short)0, c, false, false);
}

__device__ __forceinline__ uvec4 pack8_bf16(fvec4 f0, fvec4 f1) {
  uvec4 pk;
  pk.x = (unsigned)f2bf_u(f0[0]) | ((unsigned)f2bf_u(f0[1]) << 16);
  pk.y = (unsigned)f2bf_u(f0[2]) | ((unsigned)f2bf_u(f0[3]) << 16);
  pk.z = (unsigned)f2bf_u(f1[0]) | ((unsigned)f2bf_u(f1[1]) << 16);
  pk.w = (unsigned)f2bf_u(f1[2]) | ((unsigned)f2bf_u(f1[3]) << 16);
  return pk;
}

// ---------------------------------------------------------------------------
// Projection GEMM: out(bf16) = X1@W1 [+ X2@W2] + b1 [+ b2]
// 8192 x 512 x 512. Tile 64 rows x 128 cols, 8 waves (4 row-bands x 2
// col-bands), K-step 32. DUAL/TR are compile-time: branch-free inner loop.
// ---------------------------------------------------------------------------
template <bool DUAL, bool TR>
__global__ __launch_bounds__(256) void proj_kernel(
    const float* __restrict__ X1, const float* __restrict__ W1,
    const float* __restrict__ b1, const float* __restrict__ X2,
    const float* __restrict__ W2, const float* __restrict__ b2,
    unsigned short* __restrict__ out) {
  __shared__ __align__(16) unsigned short sX1[64][32];
  __shared__ __align__(16) unsigned short sX2[DUAL ? 64 : 1][32];
  __shared__ __align__(16) unsigned short sW1[128][32];
  __shared__ __align__(16) unsigned short sW2[DUAL ? 128 : 1][32];

  const int tid  = threadIdx.x;
  const int lane = tid & 31;
  const int w    = tid >> 5;
  const int wr   = w & 3;
  const int wc   = w >> 2;
  const int row0 = blockIdx.y * 64;
  const int n0   = blockIdx.x * 128;

  v8f acc[4];
#pragma unroll
  for (int j = 0; j < 4; ++j) acc[j] = vzero8();

  for (int k0 = 0; k0 < cD; k0 += 32) {
    // ---- stage X tiles: 64 rows x 32 k, fp32 -> bf16 ----
    {
      const int idx = tid * 8;
      const int r = idx >> 5, c = idx & 31;
      const float* p = X1 + (size_t)(row0 + r) * cD + k0 + c;
      *(uvec4*)&sX1[r][c] = pack8_bf16(*(const fvec4*)p, *(const fvec4*)(p + 4));
      if constexpr (DUAL) {
        const float* q = X2 + (size_t)(row0 + r) * cD + k0 + c;
        *(uvec4*)&sX2[r][c] = pack8_bf16(*(const fvec4*)q, *(const fvec4*)(q + 4));
      }
    }
    // ---- stage W tiles transposed: lds[n][k], 128 n x 32 k ----
    {
      const int kk = tid >> 3;
      const int nb = (tid & 7) * 16;
      const float* p = W1 + (size_t)(k0 + kk) * cD + n0 + nb;
#pragma unroll
      for (int i = 0; i < 16; i += 4) {
        fvec4 f = *(const fvec4*)(p + i);
        sW1[nb + i + 0][kk] = f2bf_u(f[0]);
        sW1[nb + i + 1][kk] = f2bf_u(f[1]);
        sW1[nb + i + 2][kk] = f2bf_u(f[2]);
        sW1[nb + i + 3][kk] = f2bf_u(f[3]);
      }
      if constexpr (DUAL) {
        const float* q = W2 + (size_t)(k0 + kk) * cD + n0 + nb;
#pragma unroll
        for (int i = 0; i < 16; i += 4) {
          fvec4 f = *(const fvec4*)(q + i);
          sW2[nb + i + 0][kk] = f2bf_u(f[0]);
          sW2[nb + i + 1][kk] = f2bf_u(f[1]);
          sW2[nb + i + 2][kk] = f2bf_u(f[2]);
          sW2[nb + i + 3][kk] = f2bf_u(f[3]);
        }
      }
    }
    __syncthreads();

    // ---- compute: branch-free 4 (or 8) WMMAs per wave per K-step ----
    const int ra = wr * 16 + (lane & 15);
    const int kg = (lane >> 4) * 8;
    v16bf a1 = load_bf16x16(&sX1[ra][kg], &sX1[ra][16 + kg]);
    v16bf a2;
    if constexpr (DUAL) a2 = load_bf16x16(&sX2[ra][kg], &sX2[ra][16 + kg]);
#pragma unroll
    for (int j = 0; j < 4; ++j) {
      const int rb = wc * 64 + j * 16 + (lane & 15);
      v16bf bf = load_bf16x16(&sW1[rb][kg], &sW1[rb][16 + kg]);
      acc[j] = wmma_bf(a1, bf, acc[j]);
      if constexpr (DUAL) {
        v16bf bf2 = load_bf16x16(&sW2[rb][kg], &sW2[rb][16 + kg]);
        acc[j] = wmma_bf(a2, bf2, acc[j]);
      }
    }
    __syncthreads();
  }

  // ---- epilogue: bias + head-split bf16 store ----
#pragma unroll
  for (int j = 0; j < 4; ++j) {
    const int col = n0 + wc * 64 + j * 16 + (lane & 15);
    float bias = b1[col];
    if constexpr (DUAL) bias += b2[col];
    const int h = col >> 6, dh = col & 63;
#pragma unroll
    for (int i = 0; i < 8; ++i) {
      const int tokr = row0 + wr * 16 + i + ((lane >> 4) << 3);
      const int b = tokr >> 11, tok = tokr & (cN - 1);   // L == 2048
      const float val = acc[j][i] + bias;
      size_t o;
      if constexpr (TR)
        o = ((size_t)((b * cH + h) * cDH + dh)) * cN + tok;
      else
        o = ((size_t)(b * cH + h) * cN + tok) * cDH + dh;
      out[o] = f2bf_u(val);
    }
  }
}

// ---------------------------------------------------------------------------
// Attention: one block = (b,h, 16 query rows). 256 threads = 8 waves.
// Dynamic LDS: 16x2048 fp32 scores (128KB) + 4KB PV reduction scratch.
// ---------------------------------------------------------------------------
__global__ __launch_bounds__(256) void attn_kernel(
    const unsigned short* __restrict__ qb,   // (b,h,n,dh) bf16
    const unsigned short* __restrict__ kb,   // (b,h,m,dh) bf16
    const unsigned short* __restrict__ vtb,  // (b,h,dh,m) bf16
    const float* __restrict__ af,            // (b,n,m)
    const unsigned char* __restrict__ mask,  // (b,m) bool
    float* __restrict__ out_hidden,          // (b,n,d)
    float* __restrict__ out_attn) {          // (b,h,n,m)
  extern __shared__ char smem_raw[];
  float* sc  = (float*)smem_raw;   // [16][2048] scores -> packed bf16 probs
  float* red = sc + 16 * cM;       // [4][256] PV split-K partials

  const int tid  = threadIdx.x;
  const int lane = tid & 31;
  const int w    = tid >> 5;
  const int bh   = blockIdx.y;
  const int b    = bh >> 3;
  const int h    = bh & 7;
  const int n0   = blockIdx.x * 16;
  const int rl   = lane & 15;
  const int kg   = (lane >> 4) * 8;
  const float NEG_INF = -__builtin_inff();

  // Q fragments for this 16-row tile (K = dh = 64 -> two 16x32 A frags)
  const unsigned short* qrow = qb + ((size_t)bh * cN + n0 + rl) * cDH;
  v16bf qa0 = load_bf16x16(qrow + kg,      qrow + 16 + kg);
  v16bf qa1 = load_bf16x16(qrow + 32 + kg, qrow + 48 + kg);

  const unsigned short* kbase  = kb + (size_t)bh * cM * cDH;
  const unsigned char*  mrow   = mask + (size_t)b * cM;
  const float*          afbase = af + ((size_t)b * cN + n0) * cM;

  // ---- scores: each wave owns 16 of the 128 key chunks ----
  for (int ch = w; ch < cM / 16; ch += 8) {
    const int m0 = ch * 16;
    const unsigned short* kr = kbase + (size_t)(m0 + rl) * cDH;
    v16bf kf0 = load_bf16x16(kr + kg,      kr + 16 + kg);
    v16bf kf1 = load_bf16x16(kr + 32 + kg, kr + 48 + kg);
    v8f c = vzero8();
    c = wmma_bf(qa0, kf0, c);
    c = wmma_bf(qa1, kf1, c);
    const int coll = m0 + (lane & 15);
    const bool km = (mrow[coll] != 0);
#pragma unroll
    for (int i = 0; i < 8; ++i) {
      const int r = i + ((lane >> 4) << 3);
      float s = c[i] * 0.125f * afbase[(size_t)r * cM + coll];
      s = km ? NEG_INF : s;
      sc[r * cM + coll] = s;
    }
  }
  __syncthreads();

  // ---- softmax: wave w owns rows 2w,2w+1. Writes attn (b64 stores) and
  //      re-packs probs bf16 in-place into the low half of each LDS row:
  //      write range [128i,128i+128) never overlaps later reads [256j,...).
#pragma unroll
  for (int rr = 0; rr < 2; ++rr) {
    const int r = w * 2 + rr;
    float* row = sc + (size_t)r * cM;
    float mx = NEG_INF;
    for (int m = lane; m < cM; m += 32) mx = fmaxf(mx, row[m]);
#pragma unroll
    for (int off = 16; off > 0; off >>= 1) mx = fmaxf(mx, __shfl_xor(mx, off, 32));
    float s = 0.0f;
    for (int m = lane; m < cM; m += 32) s += __expf(row[m] - mx);
#pragma unroll
    for (int off = 16; off > 0; off >>= 1) s += __shfl_xor(s, off, 32);
    const float inv = 1.0f / s;
    float* orow = out_attn + ((size_t)bh * cN + n0 + r) * cM;
    unsigned* rowbf = (unsigned*)row;   // packed bf16 pairs, element m>>1
    for (int m = lane * 2; m < cM; m += 64) {
      const float p0 = __expf(row[m] - mx) * inv;
      const float p1 = __expf(row[m + 1] - mx) * inv;
      fvec2 pp; pp[0] = p0; pp[1] = p1;
      *(fvec2*)(orow + m) = pp;
      rowbf[m >> 1] = (unsigned)f2bf_u(p0) | ((unsigned)f2bf_u(p1) << 16);
    }
  }
  __syncthreads();

  // ---- PV: 8-wave split-K. wave w -> dh tile (w&3)*16, K-half (w>>2).
  //      Inner loop: 2 ds_load_b128 + 2 global_load_b128 + 1 WMMA.
  const int dh0 = (w & 3) * 16;
  const int k0  = (w >> 2) * (cM / 2);
  const unsigned short* vt  = vtb + ((size_t)bh * cDH + dh0 + rl) * cM;
  const unsigned short* abf = (const unsigned short*)(sc + (size_t)rl * cM);
  v8f acc = vzero8();
  for (int kk = k0; kk < k0 + cM / 2; kk += 32) {
    v16bf a  = load_bf16x16(abf + kk + kg, abf + kk + 16 + kg);
    v16bf bf = load_bf16x16(vt  + kk + kg, vt  + kk + 16 + kg);
    acc = wmma_bf(a, bf, acc);
  }
  // waves 4..7 publish partials, waves 0..3 combine and store hidden
  if (w >= 4) {
#pragma unroll
    for (int i = 0; i < 8; ++i) red[(w - 4) * 256 + i * 32 + lane] = acc[i];
  }
  __syncthreads();
  if (w < 4) {
#pragma unroll
    for (int i = 0; i < 8; ++i) {
      const float v  = acc[i] + red[w * 256 + i * 32 + lane];
      const int   r  = i + ((lane >> 4) << 3);
      const int   dh = dh0 + (lane & 15);
      out_hidden[((size_t)(b * cN + n0 + r)) * cD + h * cDH + dh] = v;
    }
  }
}

// ---------------------------------------------------------------------------
extern "C" void kernel_launch(void* const* d_in, const int* in_sizes, int n_in,
                              void* d_out, int out_size, void* d_ws,
                              size_t ws_size, hipStream_t stream) {
  const float* input_q = (const float*)d_in[0];
  const float* input_k = (const float*)d_in[1];
  const float* input_v = (const float*)d_in[2];
  const float* embed_q = (const float*)d_in[3];
  const float* embed_k = (const float*)d_in[4];
  const unsigned char* key_masks = (const unsigned char*)d_in[5];
  const float* att_fac = (const float*)d_in[6];
  const float* Wq = (const float*)d_in[7];
  const float* bq = (const float*)d_in[8];
  const float* Wk = (const float*)d_in[9];
  const float* bk = (const float*)d_in[10];
  const float* Wv = (const float*)d_in[11];
  const float* bv = (const float*)d_in[12];
  const float* Wp = (const float*)d_in[13];
  const float* bp = (const float*)d_in[14];

  const size_t per_tensor = (size_t)cB * cH * cN * cDH;  // 4,194,304 bf16
  unsigned short* qws = (unsigned short*)d_ws;
  unsigned short* kws = qws + per_tensor;
  unsigned short* vws = kws + per_tensor;

  dim3 gA(cD / 128, (cB * cN) / 64), bA(256);
  proj_kernel<true, false><<<gA, bA, 0, stream>>>(input_q, Wq, bq, embed_q, Wp,
                                                  bp, qws);
  proj_kernel<true, false><<<gA, bA, 0, stream>>>(input_k, Wk, bk, embed_k, Wp,
                                                  bp, kws);
  proj_kernel<false, true><<<gA, bA, 0, stream>>>(input_v, Wv, bv, nullptr,
                                                  nullptr, nullptr, vws);

  float* out_hidden = (float*)d_out;
  float* out_attn   = out_hidden + (size_t)cB * cN * cD;

  const size_t smem = (size_t)16 * cM * sizeof(float) + 4 * 256 * sizeof(float);
  (void)hipFuncSetAttribute((const void*)attn_kernel,
                            hipFuncAttributeMaxDynamicSharedMemorySize,
                            (int)smem);
  dim3 gB(cN / 16, cB * cH), bB(256);
  attn_kernel<<<gB, bB, smem, stream>>>(qws, kws, vws, att_fac, key_masks,
                                        out_hidden, out_attn);
}
